// LSTMmodel_14516989461340
// MI455X (gfx1250) — matrix operand order repaired and checked by reading, not compiled, not run on hardware
//
#include <hip/hip_runtime.h>

typedef __attribute__((ext_vector_type(8)))  _Float16 v8h;
typedef __attribute__((ext_vector_type(16))) _Float16 v16h;
typedef __attribute__((ext_vector_type(8)))  float    v8f;

typedef unsigned int       u32;
typedef unsigned long long u64;
typedef u32 u32x4 __attribute__((ext_vector_type(4)));
typedef int i32x4 __attribute__((ext_vector_type(4)));
typedef int i32x8 __attribute__((ext_vector_type(8)));

#define T_SEQ 4096
#define D_IN  32
#define D_H   512
#define D_OUT 32
#define G4H   2048   // 4*H gate rows, PyTorch order i,f,g,o
#define NWG   16     // persistent workgroups for the recurrence

#if __has_builtin(__builtin_amdgcn_tensor_load_to_lds) && \
    __has_builtin(__builtin_amdgcn_s_wait_tensorcnt)
#define HAVE_TDM 1
#else
#define HAVE_TDM 0
#endif

// ---------------------------------------------------------------------------
// fp32 -> fp16 conversion (for activations and weights feeding WMMA)
// ---------------------------------------------------------------------------
__global__ void k_f32_to_f16(const float* __restrict__ s, _Float16* __restrict__ d, int n) {
    int i = blockIdx.x * 256 + threadIdx.x;
    if (i < n) d[i] = (_Float16)s[i];
}

__global__ void k_zero_u32(unsigned* __restrict__ p, int n) {
    int i = blockIdx.x * 256 + threadIdx.x;
    if (i < n) p[i] = 0u;
}

#if HAVE_TDM
// ---------------------------------------------------------------------------
// Tensor Data Mover: 2D tile (rows x row_len f16 elements) global -> LDS.
// D# layout per CDNA5 ISA:
//   group0: count=1, lds_addr[63:32], global_addr[120:64], type=2 [127:126]
//   group1: data_size=1 (2B) [17:16], tensor_dim0 [79:48], tensor_dim1 [111:80],
//           tile_dim0 [127:112], tile_dim1 [143:128], tensor_dim0_stride [207:160]
//   groups 2/3: zero (2D tensor)
// ---------------------------------------------------------------------------
__device__ __forceinline__ void tdm_load_2d_f16(void* lds_dst, const void* gsrc,
                                                u32 row_len, u32 rows, u32 stride) {
    u64 g0q0 = 1ull | ((u64)(u32)(size_t)lds_dst << 32);
    u64 g0q1 = ((u64)(size_t)gsrc & 0x01FFFFFFFFFFFFFFull) | (2ull << 62);
    u32x4 g0;
    g0[0] = (u32)g0q0; g0[1] = (u32)(g0q0 >> 32);
    g0[2] = (u32)g0q1; g0[3] = (u32)(g0q1 >> 32);

    u64 q0 = ((u64)1 << 16)                               // data_size = 2 bytes
           | ((u64)(row_len & 0xFFFFu) << 48);            // tensor_dim0[15:0]
    u64 q1 = ((u64)(row_len >> 16) & 0xFFFFull)           // tensor_dim0[31:16]
           | ((u64)rows << 16)                            // tensor_dim1
           | ((u64)(row_len & 0xFFFFu) << 48);            // tile_dim0
    u64 q2 = ((u64)(rows & 0xFFFFu))                      // tile_dim1
           | ((u64)stride << 32);                         // tensor_dim0_stride[31:0]
    u64 q3 = 0;                                           // stride[47:32]=0, dim1_stride=0
    i32x8 g1;
    g1[0] = (int)(u32)q0; g1[1] = (int)(u32)(q0 >> 32);
    g1[2] = (int)(u32)q1; g1[3] = (int)(u32)(q1 >> 32);
    g1[4] = (int)(u32)q2; g1[5] = (int)(u32)(q2 >> 32);
    g1[6] = (int)(u32)q3; g1[7] = (int)(u32)(q3 >> 32);
    i32x4 z = {0, 0, 0, 0};
#if defined(__clang_major__) && __clang_major__ >= 23
    i32x8 z8 = {0, 0, 0, 0, 0, 0, 0, 0};
    __builtin_amdgcn_tensor_load_to_lds(g0, g1, z, z, z8, 0);
#else
    __builtin_amdgcn_tensor_load_to_lds(g0, g1, z, z, 0);
#endif
}
#endif

// ---------------------------------------------------------------------------
// Simple WMMA GEMM (1 wave per 16x16 tile): C = A[M,K] @ W[N,K]^T + b0 (+b1).
// Kept for the final projection (N=32, not a 64-wide panel).
// ---------------------------------------------------------------------------
__global__ __launch_bounds__(32)
void k_wmma_gemm_bias(const _Float16* __restrict__ A, const _Float16* __restrict__ W,
                      const float* __restrict__ b0, const float* __restrict__ b1,
                      float* __restrict__ C, int M, int N, int K) {
    const int m0   = blockIdx.x * 16;
    const int n0   = blockIdx.y * 16;
    const int lane = threadIdx.x;
    const bool hi  = lane >= 16;
    const int  l   = lane & 15;

    const _Float16* arow = A + (size_t)(m0 + l) * K;
    const _Float16* wrow = W + (size_t)(n0 + l) * K;

    v8f acc = {};
    for (int k0 = 0; k0 < K; k0 += 32) {
        int ka = k0 + (hi ? 16 : 0);
        v8h alo = *(const v8h*)(arow + ka);
        v8h ahi = *(const v8h*)(arow + ka + 8);
        v16h af;
#pragma unroll
        for (int i = 0; i < 8; ++i) { af[i] = alo[i]; af[i + 8] = ahi[i]; }
        int kb = k0 + (hi ? 16 : 0);
        v16h bf = *(const v16h*)(wrow + kb);
        acc = __builtin_amdgcn_wmma_f32_16x16x32_f16(false, af, false, bf,
                                                     (short)0, acc, false, false);
    }
    const int col  = n0 + l;
    const float bias = (b0 ? b0[col] : 0.f) + (b1 ? b1[col] : 0.f);
    const int rbase = m0 + (hi ? 8 : 0);
#pragma unroll
    for (int r = 0; r < 8; ++r)
        C[(size_t)(rbase + r) * N + col] = acc[r] + bias;
}

// ---------------------------------------------------------------------------
// Paneled WMMA GEMM: block = 4 waves = 64x64 C block. The 64-row B panel
// (contiguous 64*K f16 block of row-major W) is staged into LDS once via the
// Tensor Data Mover, then each wave reuses one A fragment across 4 N-tiles.
// Requires: M % 64 == 0, N % 64 == 0, K % 32 == 0. Dynamic LDS = 64*K*2 B.
// ---------------------------------------------------------------------------
__global__ __launch_bounds__(128)
void k_wmma_gemm_panel(const _Float16* __restrict__ A, const _Float16* __restrict__ W,
                       const float* __restrict__ b0, const float* __restrict__ b1,
                       float* __restrict__ C, int M, int N, int K) {
    extern __shared__ char psm[];
    _Float16* wl = (_Float16*)psm;                 // [64][K] B panel

    const int m0 = blockIdx.x * 64;
    const int n0 = blockIdx.y * 64;
    const _Float16* Wp = W + (size_t)n0 * K;       // contiguous 64*K block

#if HAVE_TDM
    if (threadIdx.x < 32) {                        // wave 0 issues the DMA
        tdm_load_2d_f16(wl, Wp, (u32)K, 64u, (u32)K);
        __builtin_amdgcn_s_wait_tensorcnt(0);
    }
    __syncthreads();
#else
    for (int idx = threadIdx.x; idx < 64 * K; idx += 128) wl[idx] = Wp[idx];
    __syncthreads();
#endif

    const int lane = threadIdx.x & 31;
    const int wave = threadIdx.x >> 5;
    const bool hi  = lane >= 16;
    const int  l   = lane & 15;

    const _Float16* arow = A + (size_t)(m0 + wave * 16 + l) * K;

    v8f acc[4] = {};
    for (int k0 = 0; k0 < K; k0 += 32) {
        int ka = k0 + (hi ? 16 : 0);
        v8h alo = *(const v8h*)(arow + ka);
        v8h ahi = *(const v8h*)(arow + ka + 8);
        v16h af;
#pragma unroll
        for (int i = 0; i < 8; ++i) { af[i] = alo[i]; af[i + 8] = ahi[i]; }
        int kb = k0 + (hi ? 16 : 0);
        const _Float16* bl = wl + (size_t)l * K + kb;   // col l of each 16-col tile
#pragma unroll
        for (int nt = 0; nt < 4; ++nt) {
            v16h bf = *(const v16h*)(bl + (size_t)(nt * 16) * K);
            acc[nt] = __builtin_amdgcn_wmma_f32_16x16x32_f16(false, af, false, bf,
                                                             (short)0, acc[nt], false, false);
        }
    }
    const int rbase = m0 + wave * 16 + (hi ? 8 : 0);
#pragma unroll
    for (int nt = 0; nt < 4; ++nt) {
        int col = n0 + nt * 16 + l;
        float bias = (b0 ? b0[col] : 0.f) + (b1 ? b1[col] : 0.f);
#pragma unroll
        for (int r = 0; r < 8; ++r)
            C[(size_t)(rbase + r) * N + col] = acc[nt][r] + bias;
    }
}

// ---------------------------------------------------------------------------
// Persistent LSTM recurrence over one layer.
// 16 WGs x 256 threads. WG k owns h indices [k*32, k*32+32) -> 128 W_hh rows
// (i,f,g,o slices) cached in LDS as f16 (128 KB). Cell state c is register-
// resident in threads 0..31. h is double-buffered in global memory; a
// monotonic-counter device barrier syncs the 16 WGs each timestep.
// G = precomputed x@W_ih^T + b_ih + b_hh  [T, 4H] f32.
// ---------------------------------------------------------------------------
__global__ __launch_bounds__(256)
void k_lstm_rec(const float* __restrict__ G, const float* __restrict__ Whh,
                float* __restrict__ h_pp,      // [2][512] double-buffered h, zero-init
                _Float16* __restrict__ h_hist, // [T, 512] f16 output history
                unsigned* __restrict__ counter) {
    extern __shared__ char smem[];
    _Float16* wl = (_Float16*)smem;                       // 128*512 f16
    float*    hl = (float*)(smem + 128 * 512 * 2);        // 512 f32 h_prev
    float*    gl = (float*)(smem + 128 * 512 * 2 + 2048); // 128 gate pre-acts

    const int wg  = blockIdx.x;    // 0..15
    const int tid = threadIdx.x;   // 0..255
    const int j0  = wg * 32;       // first h index owned

    for (int idx = tid; idx < 128 * 512; idx += 256) {
        int r = idx >> 9, k = idx & 511;
        int q = r >> 5, jj = r & 31;
        wl[idx] = (_Float16)Whh[(size_t)(q * 512 + j0 + jj) * 512 + k];
    }
    float c = 0.f;  // cell state for h index j0+tid (threads 0..31)
    __syncthreads();

    const int r    = tid >> 1;          // gate row 0..127
    const int half = tid & 1;           // which 256-element half of the dot
    const int q    = r >> 5, jj = r & 31;
    const size_t grow = (size_t)q * 512 + j0 + jj;
    const _Float16* wrow = wl + r * 512 + half * 256;

    for (int t = 0; t < T_SEQ; ++t) {
        const float* hprev = h_pp + (t & 1) * 512;
        float*       hnext = h_pp + ((t + 1) & 1) * 512;

        for (int i = tid; i < 512; i += 256) hl[i] = hprev[i];
        __syncthreads();

        if (t + 1 < T_SEQ)
            __builtin_prefetch(G + (size_t)(t + 1) * G4H + grow, 0, 1);

        const float* hseg = hl + half * 256;
        float acc = 0.f;
#pragma unroll 8
        for (int k = 0; k < 256; ++k) acc += (float)wrow[k] * hseg[k];
        acc += __shfl_xor(acc, 1, 32);
        if (half == 0) gl[r] = acc + G[(size_t)t * G4H + grow];
        __syncthreads();

        if (tid < 32) {
            float gi = gl[tid], gf = gl[32 + tid], gg = gl[64 + tid], go = gl[96 + tid];
            float i_ = 1.f / (1.f + __expf(-gi));
            float f_ = 1.f / (1.f + __expf(-gf));
            float g_ = tanhf(gg);
            float o_ = 1.f / (1.f + __expf(-go));
            c = f_ * c + i_ * g_;
            float h = o_ * tanhf(c);
            hnext[j0 + tid] = h;
            h_hist[(size_t)t * D_H + j0 + tid] = (_Float16)h;
        }
        __syncthreads();
        if (tid == 0) {
            __threadfence();
            atomicAdd(counter, 1u);
            unsigned target = (unsigned)(t + 1) * NWG;
            while (__hip_atomic_load(counter, __ATOMIC_ACQUIRE,
                                     __HIP_MEMORY_SCOPE_AGENT) < target)
                __builtin_amdgcn_s_sleep(1);
        }
        __syncthreads();
    }
}

// ---------------------------------------------------------------------------
extern "C" void kernel_launch(void* const* d_in, const int* in_sizes, int n_in,
                              void* d_out, int out_size, void* d_ws, size_t ws_size,
                              hipStream_t stream) {
    (void)in_sizes; (void)n_in; (void)out_size; (void)ws_size;
    const float* seq   = (const float*)d_in[0];
    const float* W_ih1 = (const float*)d_in[1];
    const float* W_hh1 = (const float*)d_in[2];
    const float* b_ih1 = (const float*)d_in[3];
    const float* b_hh1 = (const float*)d_in[4];
    const float* W_ih2 = (const float*)d_in[5];
    const float* W_hh2 = (const float*)d_in[6];
    const float* b_ih2 = (const float*)d_in[7];
    const float* b_hh2 = (const float*)d_in[8];
    const float* W_ih3 = (const float*)d_in[9];
    const float* W_hh3 = (const float*)d_in[10];
    const float* b_ih3 = (const float*)d_in[11];
    const float* b_hh3 = (const float*)d_in[12];
    const float* W_out = (const float*)d_in[13];
    const float* b_out = (const float*)d_in[14];
    float* out = (float*)d_out;

    char* ws = (char*)d_ws;
    size_t off = 0;
    float*    g_buf  = (float*)(ws + off);     off += (size_t)T_SEQ * G4H * 4;   // 32 MB gates
    _Float16* h_hist = (_Float16*)(ws + off);  off += (size_t)T_SEQ * D_H * 2;   // 4 MB hidden history
    _Float16* seq16  = (_Float16*)(ws + off);  off += (size_t)T_SEQ * D_IN * 2;
    _Float16* wbuf   = (_Float16*)(ws + off);  off += (size_t)G4H * D_H * 2;     // 2 MB f16 weights
    float*    h_pp   = (float*)(ws + off);     off += 1024 * 4;                  // h double buffer
    unsigned* counter = (unsigned*)(ws + off); off += 256;
    const int n_state_words = 1024 + 64;       // h_pp + counter region, contiguous

    const int rec_smem = 128 * 512 * 2 + 2048 + 512;
    hipFuncSetAttribute((const void*)k_lstm_rec,
                        hipFuncAttributeMaxDynamicSharedMemorySize, rec_smem);
    const int panel_smem_big = 64 * D_H * 2;   // 64 KB for K=512
    hipFuncSetAttribute((const void*)k_wmma_gemm_panel,
                        hipFuncAttributeMaxDynamicSharedMemorySize, panel_smem_big);

    k_f32_to_f16<<<dim3((T_SEQ * D_IN + 255) / 256), 256, 0, stream>>>(seq, seq16, T_SEQ * D_IN);

    // ---- layer 1 (K = 32) ----
    k_f32_to_f16<<<dim3((G4H * D_IN + 255) / 256), 256, 0, stream>>>(W_ih1, wbuf, G4H * D_IN);
    k_wmma_gemm_panel<<<dim3(T_SEQ / 64, G4H / 64), 128, 64 * D_IN * 2, stream>>>(
        seq16, wbuf, b_ih1, b_hh1, g_buf, T_SEQ, G4H, D_IN);
    k_zero_u32<<<dim3((n_state_words + 255) / 256), 256, 0, stream>>>((unsigned*)h_pp, n_state_words);
    k_lstm_rec<<<NWG, 256, rec_smem, stream>>>(g_buf, W_hh1, h_pp, h_hist, counter);

    // ---- layer 2 (K = 512) ----
    k_f32_to_f16<<<dim3((G4H * D_H + 255) / 256), 256, 0, stream>>>(W_ih2, wbuf, G4H * D_H);
    k_wmma_gemm_panel<<<dim3(T_SEQ / 64, G4H / 64), 128, panel_smem_big, stream>>>(
        h_hist, wbuf, b_ih2, b_hh2, g_buf, T_SEQ, G4H, D_H);
    k_zero_u32<<<dim3((n_state_words + 255) / 256), 256, 0, stream>>>((unsigned*)h_pp, n_state_words);
    k_lstm_rec<<<NWG, 256, rec_smem, stream>>>(g_buf, W_hh2, h_pp, h_hist, counter);

    // ---- layer 3 (K = 512) ----
    k_f32_to_f16<<<dim3((G4H * D_H + 255) / 256), 256, 0, stream>>>(W_ih3, wbuf, G4H * D_H);
    k_wmma_gemm_panel<<<dim3(T_SEQ / 64, G4H / 64), 128, panel_smem_big, stream>>>(
        h_hist, wbuf, b_ih3, b_hh3, g_buf, T_SEQ, G4H, D_H);
    k_zero_u32<<<dim3((n_state_words + 255) / 256), 256, 0, stream>>>((unsigned*)h_pp, n_state_words);
    k_lstm_rec<<<NWG, 256, rec_smem, stream>>>(g_buf, W_hh3, h_pp, h_hist, counter);

    // ---- output projection: [4096,512] @ [512,32] + b_out -> d_out ----
    k_f32_to_f16<<<dim3((D_OUT * D_H + 255) / 256), 256, 0, stream>>>(W_out, wbuf, D_OUT * D_H);
    k_wmma_gemm_bias<<<dim3(T_SEQ / 16, D_OUT / 16), 32, 0, stream>>>(
        h_hist, wbuf, b_out, nullptr, out, T_SEQ, D_OUT, D_H);
}
